// ALSHConv2d_66838281060970
// MI455X (gfx1250) — matrix operand order repaired and checked by reading, not compiled, not run on hardware
//
#include <hip/hip_runtime.h>
#include <hip/hip_bf16.h>

// ---------------- problem constants (from reference) ----------------
#define K_SZ   3
#define IN_C   256
#define OUT_C  512
#define HW     56
#define SPATIAL (HW*HW)           // 3136
#define NIMG   16
#define NCOL   (NIMG*SPATIAL)     // 50176
#define DDIM   (K_SZ*K_SZ*IN_C)   // 2304
#define MHASH  5

typedef __attribute__((ext_vector_type(16))) __bf16 v16bf;
typedef __attribute__((ext_vector_type(8)))  __bf16 v8bf;
typedef __attribute__((ext_vector_type(8)))  float  v8f;

// =====================================================================
// 1a) Per-channel region maxes of the im2col rows.
// =====================================================================
__global__ void regmax_kernel(const float* __restrict__ x,
                              float* __restrict__ regmax) {
  int c = blockIdx.x;
  int tid = threadIdx.x;
  float lm[9];
#pragma unroll
  for (int j = 0; j < 9; ++j) lm[j] = -INFINITY;

  for (int idx = tid; idx < NIMG * SPATIAL; idx += 256) {
    int n = idx / SPATIAL;
    int r = idx - n * SPATIAL;
    int y = r / HW;
    int xx = r - y * HW;
    float v = x[((size_t)(n * IN_C + c) * HW + y) * HW + xx];
    bool rok[3] = { y <= HW - 2, true, y >= 1 };
    bool cok[3] = { xx <= HW - 2, true, xx >= 1 };
#pragma unroll
    for (int kh = 0; kh < 3; ++kh)
#pragma unroll
      for (int kw = 0; kw < 3; ++kw)
        if (rok[kh] && cok[kw]) lm[kh*3+kw] = fmaxf(lm[kh*3+kw], v);
  }

  __shared__ float red[256];
  for (int j = 0; j < 9; ++j) {
    red[tid] = lm[j];
    __syncthreads();
    for (int s = 128; s > 0; s >>= 1) {
      if (tid < s) red[tid] = fmaxf(red[tid], red[tid + s]);
      __syncthreads();
    }
    if (tid == 0) {
      float v = red[0];
      if (j != 4) v = fmaxf(v, 0.0f);   // non-center taps see padding zeros
      regmax[j * IN_C + c] = v;         // == max_row[j*256 + c]
    }
    __syncthreads();
  }
}

// =====================================================================
// 1b) ALSH hash per kernel row: floor(dot(a, P(k))) mod 16
// =====================================================================
__global__ void khash_kernel(const float* __restrict__ kernels,
                             const float* __restrict__ a,
                             int* __restrict__ kidx) {
  int o = blockIdx.x, tid = threadIdx.x;
  const float* kr = kernels + (size_t)o * DDIM;
  float sd = 0.f, sn = 0.f;
#pragma unroll
  for (int i = 0; i < DDIM / 256; ++i) {
    float v = kr[tid + i * 256];
    sd += a[tid + i * 256] * v;
    sn += v * v;
  }
  __shared__ float r1[256], r2[256];
  r1[tid] = sd; r2[tid] = sn;
  __syncthreads();
  for (int s = 128; s > 0; s >>= 1) {
    if (tid < s) { r1[tid] += r1[tid + s]; r2[tid] += r2[tid + s]; }
    __syncthreads();
  }
  if (tid == 0) {
    float acc = r1[0], p = r2[0];
#pragma unroll
    for (int i = 0; i < MHASH; ++i) { acc += a[DDIM + i] * p; p = p * p; }
    float m = fmodf(floorf(acc), 16.0f);
    if (m < 0.f) m += 16.0f;
    kidx[o] = (int)m;
  }
}

// =====================================================================
// 1c) Query hash + routed-row multipliers (single block)
// =====================================================================
__global__ void route_kernel(const float* __restrict__ regmax,
                             const float* __restrict__ a,
                             const int* __restrict__ kidx,
                             float* __restrict__ mult) {
  int tid = threadIdx.x;
  float sn = 0.f, sd = 0.f;
#pragma unroll
  for (int i = 0; i < DDIM / 256; ++i) {
    float v = regmax[tid + i * 256];
    sn += v * v;
    sd += a[tid + i * 256] * v;
  }
  __shared__ float r1[256], r2[256];
  __shared__ int redi[256];
  __shared__ int sq, scnt;
  r1[tid] = sd; r2[tid] = sn;
  __syncthreads();
  for (int s = 128; s > 0; s >>= 1) {
    if (tid < s) { r1[tid] += r1[tid + s]; r2[tid] += r2[tid + s]; }
    __syncthreads();
  }
  if (tid == 0) {
    float norm = sqrtf(r2[0]);
    float acc = (norm > 0.f) ? (r1[0] / norm) : 0.f;
    float hs = 0.f;
    for (int i = 0; i < MHASH; ++i) hs += a[DDIM + i];
    acc += 0.5f * hs;
    float m = fmodf(floorf(acc), 16.0f);
    if (m < 0.f) m += 16.0f;
    sq = (int)m;
  }
  __syncthreads();
  int q = sq;
  redi[tid] = (kidx[tid] == q) + (kidx[tid + 256] == q);
  __syncthreads();
  for (int s = 128; s > 0; s >>= 1) {
    if (tid < s) redi[tid] += redi[tid + s];
    __syncthreads();
  }
  if (tid == 0) scnt = redi[0];
  __syncthreads();
  int count = scnt;
#pragma unroll
  for (int i = 0; i < 2; ++i) {
    int o = tid + i * 256;
    float mv;
    if (count > 0) mv = (kidx[o] == q) ? ((float)OUT_C / (float)count) : 0.f;
    else           mv = 1.f;
    mult[o] = mv;
  }
}

// =====================================================================
// 2) Implicit-im2col GEMM, bf16 WMMA / fp32 accum.
//    Block 128(M) x 64(N), K-step 64, double-buffered LDS.
//    8 waves (4x2), wave tile 32x32 -> 8 WMMAs per K-step per wave.
// =====================================================================
#define BM 128
#define BN 64
#define KB 64
#define NKIT (DDIM / KB)   // 36
#define ASTR 72            // LDS row stride (halves): 144B = 9*16B, bank-spread
#define BSTR 72

__global__ __launch_bounds__(256)
void gemm_kernel(const float* __restrict__ xin,
                 const float* __restrict__ kernels,
                 const float* __restrict__ bias,
                 const float* __restrict__ mult,
                 float* __restrict__ out) {
  __shared__ __align__(16) __bf16 A_lds[2][BM * ASTR];  // 2 x 18 KB
  __shared__ __align__(16) __bf16 B_lds[2][BN * BSTR];  // 2 x 9 KB
  __shared__ int colmeta[BN];

  int tid = threadIdx.x;
  int blockN = blockIdx.x;        // 0..783
  int blockM = blockIdx.y;        // 0..3
  int lane = tid & 31;
  int w = tid >> 5;               // wave 0..7 (wave32)
  int wm = w & 3, wn = w >> 2;
  int laneM = lane & 15, hi = lane >> 4;

  if (tid < BN) {
    int col = blockN * BN + tid;  // 3136 % 64 == 0: never crosses images
    int n = col / SPATIAL;
    int r = col - n * SPATIAL;
    int y = r / HW;
    int xx = r - y * HW;
    colmeta[tid] = (n << 12) | (y << 6) | xx;
  }
  __syncthreads();

  // ---- staging thread mappings ----
  int ar = tid >> 1;                 // A: 0..127 row, 32 contiguous k
  int ak = (tid & 1) * 32;
  const float* abase = kernels + (size_t)(blockM * BM + ar) * DDIM + ak;

  int bcol = tid & 63;               // B: 1 col, 16 consecutive k
  int bk = (tid >> 6) * 16;
  int bmeta = colmeta[bcol];
  int bn_ = bmeta >> 12, by = (bmeta >> 6) & 63, bx = bmeta & 63;

  float4 av[8];                      // staged A (32 floats)
  float  bv[16];                     // staged B

  auto loadA = [&](int k0) {
    const float4* src = (const float4*)(abase + k0);
#pragma unroll
    for (int i = 0; i < 8; ++i) av[i] = src[i];
  };
  auto loadB = [&](int k0) {
    int k = k0 + bk;                 // 16-aligned: one (kh,kw) for the run
    int c0 = k & (IN_C - 1);
    int kwh = k >> 8;                // 0..8
    int kh = (kwh >= 6) ? 2 : ((kwh >= 3) ? 1 : 0);
    int kw = kwh - kh * 3;
    int py = by + kh - 1, px = bx + kw - 1;
    bool valid = ((unsigned)py < (unsigned)HW) && ((unsigned)px < (unsigned)HW);
    const float* src = xin + (((size_t)(bn_ * IN_C + c0) * HW + py) * HW + px);
#pragma unroll
    for (int i = 0; i < 16; ++i) bv[i] = valid ? src[(size_t)i * SPATIAL] : 0.0f;
  };
  auto storeA = [&](int buf) {
    __bf16* dst = &A_lds[buf][ar * ASTR + ak];
#pragma unroll
    for (int i = 0; i < 4; ++i) {
      float4 u = av[2 * i], v = av[2 * i + 1];
      v8bf p = { (__bf16)u.x, (__bf16)u.y, (__bf16)u.z, (__bf16)u.w,
                 (__bf16)v.x, (__bf16)v.y, (__bf16)v.z, (__bf16)v.w };
      *(v8bf*)(dst + i * 8) = p;
    }
  };
  auto storeB = [&](int buf) {
    __bf16* dst = &B_lds[buf][bcol * BSTR + bk];
#pragma unroll
    for (int i = 0; i < 2; ++i) {
      v8bf p = { (__bf16)bv[8*i+0], (__bf16)bv[8*i+1], (__bf16)bv[8*i+2], (__bf16)bv[8*i+3],
                 (__bf16)bv[8*i+4], (__bf16)bv[8*i+5], (__bf16)bv[8*i+6], (__bf16)bv[8*i+7] };
      *(v8bf*)(dst + i * 8) = p;
    }
  };

  v8f acc[2][2];
  v8f zero = {0.f,0.f,0.f,0.f,0.f,0.f,0.f,0.f};
#pragma unroll
  for (int i = 0; i < 2; ++i)
#pragma unroll
    for (int j = 0; j < 2; ++j) acc[i][j] = zero;

  // ---- software pipeline: prologue ----
  loadA(0); loadB(0);
  storeA(0); storeB(0);
  __syncthreads();

  for (int it = 0; it < NKIT; ++it) {
    int cur = it & 1;
    bool more = (it + 1 < NKIT);
    if (more) {                        // issue next tile's global loads early
      loadA((it + 1) * KB);
      loadB((it + 1) * KB);
      if (it + 2 < NKIT) __builtin_prefetch(abase + (it + 2) * KB, 0, 3);
    }

    // ---- 2 x (K=32) WMMA chunks from LDS[cur] ----
#pragma unroll
    for (int kc = 0; kc < 2; ++kc) {
      v16bf afrag[2], bfrag[2];
#pragma unroll
      for (int sm = 0; sm < 2; ++sm) {
        int row = wm * 32 + sm * 16 + laneM;
        int base = row * ASTR + kc * 32;
        // CDNA5 16-bit A 16x32: lanes 0-15 K{0..7,16..23}; lanes 16-31 K{8..15,24..31}
        v8bf a0 = *(const v8bf*)&A_lds[cur][base + hi * 8];
        v8bf a1 = *(const v8bf*)&A_lds[cur][base + 16 + hi * 8];
        afrag[sm] = __builtin_shufflevector(a0, a1,
            0,1,2,3,4,5,6,7,8,9,10,11,12,13,14,15);
      }
#pragma unroll
      for (int sn = 0; sn < 2; ++sn) {
        int ccol = wn * 32 + sn * 16 + laneM;
        int base = ccol * BSTR + kc * 32;
        // CDNA5 16-bit B 32x16: lanes 0-15 K{0..7,16..23}; lanes 16-31 K{8..15,24..31} (N = lane%16)
        v8bf b0 = *(const v8bf*)&B_lds[cur][base + hi * 8];
        v8bf b1 = *(const v8bf*)&B_lds[cur][base + 16 + hi * 8];
        bfrag[sn] = __builtin_shufflevector(b0, b1,
            0,1,2,3,4,5,6,7,8,9,10,11,12,13,14,15);
      }
#pragma unroll
      for (int sm = 0; sm < 2; ++sm)
#pragma unroll
        for (int sn = 0; sn < 2; ++sn)
          acc[sm][sn] = __builtin_amdgcn_wmma_f32_16x16x32_bf16(
              false, afrag[sm], false, bfrag[sn],
              (short)0, acc[sm][sn], false, false);
    }

    if (more) { storeA(1 - cur); storeB(1 - cur); }  // fill the other buffer
    __syncthreads();
  }

  // ---- epilogue: routed multiplier + bias, NCHW store ----
  int colBase = blockN * BN;
#pragma unroll
  for (int sm = 0; sm < 2; ++sm) {
    int obase = blockM * BM + wm * 32 + sm * 16 + hi * 8;  // C/D: VGPR v -> M = v + 8*hi
    float mv[8], bvv[8];
#pragma unroll
    for (int v = 0; v < 8; ++v) { mv[v] = mult[obase + v]; bvv[v] = bias[obase + v]; }
#pragma unroll
    for (int sn = 0; sn < 2; ++sn) {
      int col = colBase + wn * 32 + sn * 16 + laneM;       // C/D: N = lane%16
      int n = col / SPATIAL;
      int s = col - n * SPATIAL;
      float* op = out + (size_t)n * ((size_t)OUT_C * SPATIAL) + s;
#pragma unroll
      for (int v = 0; v < 8; ++v) {
        op[(size_t)(obase + v) * SPATIAL] = acc[sm][sn][v] * mv[v] + bvv[v];
      }
    }
  }
}

// =====================================================================
extern "C" void kernel_launch(void* const* d_in, const int* in_sizes, int n_in,
                              void* d_out, int out_size, void* d_ws, size_t ws_size,
                              hipStream_t stream) {
  const float* x       = (const float*)d_in[0];
  const float* kernels = (const float*)d_in[1];
  const float* bias    = (const float*)d_in[2];
  const float* a       = (const float*)d_in[3];
  // d_in[4] = mode (unused by the reference math)
  float* out = (float*)d_out;

  // workspace: regmax[2304] | kidx[512] (int) | mult[512]  (~13.3 KB)
  float* wsf    = (float*)d_ws;
  float* regmax = wsf;
  int*   kidx   = (int*)(wsf + DDIM);
  float* mult   = wsf + DDIM + OUT_C;

  regmax_kernel<<<IN_C, 256, 0, stream>>>(x, regmax);
  khash_kernel <<<OUT_C, 256, 0, stream>>>(kernels, a, kidx);
  route_kernel <<<1, 256, 0, stream>>>(regmax, a, kidx, mult);
  gemm_kernel  <<<dim3(NCOL / BN, OUT_C / BM), 256, 0, stream>>>(
      x, kernels, bias, mult, out);
}
